// DeformableTransformer_34806414967061
// MI455X (gfx1250) — matrix operand order repaired and checked by reading, not compiled, not run on hardware
//
#include <hip/hip_runtime.h>

// ---------------------------------------------------------------------------
// Deformable transformer encoder (3 layers) for gfx1250 (MI455X).
// GEMMs: v_wmma_f32_16x16x32_bf16, bf16 operands staged to LDS with
// global_load_async_to_lds_b128 (ASYNCcnt), fp32 accumulate.
// Residuals / LayerNorm / softmax / bilinear sampling stay fp32; every
// GEMM-consumed activation has a bf16 shadow written by its producer.
// ---------------------------------------------------------------------------

#ifndef USE_ASYNC_LDS
#define USE_ASYNC_LDS 1
#endif

typedef __bf16 bf16x16 __attribute__((ext_vector_type(16)));
typedef float  f32x8   __attribute__((ext_vector_type(8)));

constexpr int       kBatch = 2;
constexpr int       kC     = 256;
constexpr int       kHeads = 8;
constexpr int       kCh    = 32;
constexpr int       kL     = 20197;           // 15200 + 3800 + 950 + 247
constexpr int       kM     = kBatch * kL;     // 40394 rows for all GEMMs
constexpr long long kBLC   = (long long)kBatch * kL * kC;

union FragU {
  uint4   q2[2];
  bf16x16 bf;
};

// gfx1250 async global -> LDS copy, 16 bytes per lane, tracked by ASYNCcnt.
static __device__ __forceinline__ void async_copy16(unsigned ldsOff,
                                                    const void* g) {
  asm volatile("global_load_async_to_lds_b128 %0, %1, off"
               :: "v"(ldsOff), "v"((unsigned long long)g) : "memory");
}
static __device__ __forceinline__ void wait_async0() {
  asm volatile("s_wait_asynccnt 0x0" ::: "memory");
}

// ---------------------------------------------------------------------------
// GEMM: out[M,N] = act(A[M,K](bf16) * W[N,K](bf16)^T + bias).
// Block = 128 threads (4 waves). Block tile 64x64, K-step 32.
// outF (fp32) and/or outB (bf16) destinations, optional ReLU.
// ---------------------------------------------------------------------------
__global__ __launch_bounds__(128)
void gemm_bf16_wmma(const __bf16* __restrict__ A, const __bf16* __restrict__ W,
                    const float* __restrict__ bias,
                    float* __restrict__ outF, __bf16* __restrict__ outB,
                    int Mrows, int N, int K, int doRelu) {
  __shared__ __align__(16) unsigned short As[64 * 32];
  __shared__ __align__(16) unsigned short Bs[64 * 32];

  const int t    = threadIdx.x;
  const int wave = t >> 5;
  const int lane = t & 31;
  const int hf   = lane >> 4;   // which 16-lane half
  const int l16  = lane & 15;

  const int mBlock = blockIdx.x * 64;
  const int nBlock = blockIdx.y * 64;

#if USE_ASYNC_LDS
  const unsigned asBase = (unsigned)(size_t)&As[0];
  const unsigned bsBase = (unsigned)(size_t)&Bs[0];
#endif

  f32x8 acc[4] = {};

  for (int k0 = 0; k0 < K; k0 += 32) {
    // Stage 64x32 bf16 tiles of A and W into LDS (16B chunks; tail rows of A
    // are clamped to a valid row so EXEC stays all-ones for the async ops;
    // those rows' results are discarded by the M-guard in the epilogue).
#pragma unroll
    for (int i = 0; i < 2; ++i) {
      int id  = i * 128 + t;        // 0..255 chunk id
      int row = id >> 2;            // 0..63
      int ce  = (id & 3) * 8;       // bf16 element offset within row
      int gr  = mBlock + row;
      if (gr >= Mrows) gr = Mrows - 1;
      const __bf16* gA = A + (size_t)gr * K + k0 + ce;
      const __bf16* gB = W + (size_t)(nBlock + row) * K + k0 + ce;
#if USE_ASYNC_LDS
      async_copy16(asBase + (unsigned)(row * 32 + ce) * 2, gA);
      async_copy16(bsBase + (unsigned)(row * 32 + ce) * 2, gB);
#else
      *(uint4*)&As[row * 32 + ce] = *(const uint4*)gA;
      *(uint4*)&Bs[row * 32 + ce] = *(const uint4*)gB;
#endif
    }
#if USE_ASYNC_LDS
    wait_async0();
#endif
    __syncthreads();

    // A fragment (ISA 16-bit A 16x32 layout): lanes0-15 K{0-7,16-23},
    // lanes16-31 K{8-15,24-31}, row M = lane&15.
    FragU fa;
    {
      int am = wave * 16 + l16;
      fa.q2[0] = *(const uint4*)&As[am * 32 + 8 * hf];
      fa.q2[1] = *(const uint4*)&As[am * 32 + 16 + 8 * hf];
    }
#pragma unroll
    for (int j = 0; j < 4; ++j) {
      // B fragment: column N = lane&15, lanes0-15 K{0-7,16-23}, lanes16-31 K{8-15,24-31}.
      FragU fb;
      const unsigned short* bp = &Bs[(j * 16 + l16) * 32 + 8 * hf];
      fb.q2[0] = *(const uint4*)(bp);
      fb.q2[1] = *(const uint4*)(bp + 16);
      acc[j] = __builtin_amdgcn_wmma_f32_16x16x32_bf16(
          false, fa.bf, false, fb.bf, (short)0, acc[j], false, false);
    }
    __syncthreads();
  }

  // C layout: lane&15 = column; VGPR r -> row r + 8*(lane>=16).
  const int mRow0 = mBlock + wave * 16 + 8 * hf;
#pragma unroll
  for (int j = 0; j < 4; ++j) {
    int col = nBlock + j * 16 + l16;
    float bv = bias ? bias[col] : 0.0f;
#pragma unroll
    for (int r = 0; r < 8; ++r) {
      int row = mRow0 + r;
      if (row < Mrows) {
        float v = acc[j][r] + bv;
        if (doRelu) v = fmaxf(v, 0.0f);
        if (outF) outF[(size_t)row * N + col] = v;
        if (outB) outB[(size_t)row * N + col] = (__bf16)v;
      }
    }
  }
}

// ---------------------------------------------------------------------------
// fp32 -> bf16 weight conversion (once per launch; weights are tiny).
// ---------------------------------------------------------------------------
__global__ void cvt_bf16_kernel(const float* __restrict__ src,
                                __bf16* __restrict__ dst, long long n) {
  long long i = (long long)blockIdx.x * blockDim.x + threadIdx.x;
  if (i < n) dst[i] = (__bf16)src[i];
}

// ---------------------------------------------------------------------------
// Pack (B,C,H,W) pyramids -> (B, L, C) token-major layout.
// ---------------------------------------------------------------------------
static __device__ __forceinline__ void decode_level(int l, int& lv, int& n,
                                                    int& H, int& Wd) {
  if (l < 15200)      { lv = 0; n = l;         H = 100; Wd = 152; }
  else if (l < 19000) { lv = 1; n = l - 15200; H = 50;  Wd = 76;  }
  else if (l < 19950) { lv = 2; n = l - 19000; H = 25;  Wd = 38;  }
  else                { lv = 3; n = l - 19950; H = 13;  Wd = 19;  }
}

__global__ void pack_q_kernel(const float* __restrict__ q0, const float* __restrict__ q1,
                              const float* __restrict__ q2, const float* __restrict__ q3,
                              float* __restrict__ out, __bf16* __restrict__ out16,
                              long long total) {
  long long i = (long long)blockIdx.x * blockDim.x + threadIdx.x;
  if (i >= total) return;
  int c = (int)(i % kC);
  long long r = i / kC;
  int l = (int)(r % kL);
  int b = (int)(r / kL);
  int lv, n, H, Wd;
  decode_level(l, lv, n, H, Wd);
  const float* src = (lv == 0) ? q0 : (lv == 1) ? q1 : (lv == 2) ? q2 : q3;
  float v = src[(size_t)(b * kC + c) * H * Wd + n];
  out[i]   = v;
  out16[i] = (__bf16)v;
}

// kv tensors are only ever GEMM value-projection inputs -> bf16 shadows only.
__global__ void pack_kv_kernel(const float* __restrict__ k0, const float* __restrict__ k1,
                               const float* __restrict__ k2, const float* __restrict__ k3,
                               const float* __restrict__ le,
                               __bf16* __restrict__ out1, __bf16* __restrict__ out2,
                               long long total) {
  long long i = (long long)blockIdx.x * blockDim.x + threadIdx.x;
  if (i >= total) return;
  int c = (int)(i % kC);
  long long r = i / kC;
  int l = (int)(r % kL);
  int b = (int)(r / kL);
  int lv, n, H, Wd;
  decode_level(l, lv, n, H, Wd);
  const float* src = (lv == 0) ? k0 : (lv == 1) ? k1 : (lv == 2) ? k2 : k3;
  float pos = le[lv * kC + c];
  size_t plane = (size_t)kC * H * Wd;
  size_t base  = (size_t)b * 2 * plane + (size_t)c * H * Wd + n;
  out1[i] = (__bf16)(src[base] + pos);          // kv[:,0] + level_embed
  out2[i] = (__bf16)(src[base + plane] + pos);  // kv[:,1] + level_embed
}

// ---------------------------------------------------------------------------
// Softmax over groups of 16 (attention weights per (b, token, head)).
// ---------------------------------------------------------------------------
__global__ void softmax16_kernel(float* __restrict__ aw, int groups) {
  int g = blockIdx.x * blockDim.x + threadIdx.x;
  if (g >= groups) return;
  float* p = aw + (size_t)g * 16;
  float v[16], mx = -1e30f;
#pragma unroll
  for (int i = 0; i < 16; ++i) { v[i] = p[i]; mx = fmaxf(mx, v[i]); }
  float s = 0.f;
#pragma unroll
  for (int i = 0; i < 16; ++i) { v[i] = __expf(v[i] - mx); s += v[i]; }
  float inv = 1.0f / s;
#pragma unroll
  for (int i = 0; i < 16; ++i) p[i] = v[i] * inv;
}

// ---------------------------------------------------------------------------
// Deformable bilinear sampling: one wave per (b, token, head), lanes = 32 ch.
// Output feeds only the o-projection GEMM -> bf16.
// ---------------------------------------------------------------------------
__global__ __launch_bounds__(256)
void msda_sample_kernel(const float* __restrict__ val, const float* __restrict__ so,
                        const float* __restrict__ attn, __bf16* __restrict__ out) {
  long long gid = (long long)blockIdx.x * blockDim.x + threadIdx.x;
  int lane = (int)(gid & 31);
  long long wv = gid >> 5;
  if (wv >= (long long)kBatch * kL * kHeads) return;
  int head = (int)(wv % kHeads);
  long long r = wv / kHeads;
  int l = (int)(r % kL);
  int b = (int)(r / kL);

  // Reference point (same for all 4 levels) from the query's own pyramid cell.
  int lvq, n, Hq, Wq;
  decode_level(l, lvq, n, Hq, Wq);
  float rx = ((n % Wq) + 0.5f) / (float)Wq;
  float ry = ((n / Wq) + 0.5f) / (float)Hq;

  const long long tok = (long long)b * kL + l;
  const float* soP = so   + tok * 512 + head * 32;  // (lvl,pt,xy)
  const float* atP = attn + tok * 128 + head * 16;  // (lvl,pt)

  float acc = 0.f;
#pragma unroll
  for (int lvl = 0; lvl < 4; ++lvl) {
    const int Hs[4] = {100, 50, 25, 13};
    const int Ws[4] = {152, 76, 38, 19};
    const int St[4] = {0, 15200, 19000, 19950};
    int H = Hs[lvl], Wd = Ws[lvl], st = St[lvl];
    const float* vbase = val + ((long long)b * kL + st) * kC + head * kCh + lane;
#pragma unroll
    for (int pt = 0; pt < 4; ++pt) {
      float ox = soP[(lvl * 4 + pt) * 2 + 0];
      float oy = soP[(lvl * 4 + pt) * 2 + 1];
      float a  = atP[lvl * 4 + pt];
      float x = (rx + ox / (float)Wd) * (float)Wd - 0.5f;
      float y = (ry + oy / (float)H)  * (float)H  - 0.5f;
      float xf = floorf(x), yf = floorf(y);
      float wx = x - xf, wy = y - yf;
      int x0 = (int)xf, y0 = (int)yf;
      float s = 0.f;
#pragma unroll
      for (int dy = 0; dy < 2; ++dy) {
#pragma unroll
        for (int dx = 0; dx < 2; ++dx) {
          int xx = x0 + dx, yy = y0 + dy;
          if (xx >= 0 && xx < Wd && yy >= 0 && yy < H) {
            float wgt = (dx ? wx : 1.f - wx) * (dy ? wy : 1.f - wy);
            s += wgt * vbase[(long long)(yy * Wd + xx) * kC];
          }
        }
      }
      acc += a * s;
    }
  }
  out[tok * kC + head * kCh + lane] = (__bf16)acc;
}

// ---------------------------------------------------------------------------
// out = LayerNorm(a (+ bres)) * w + bias ; one wave per 256-wide row.
// Optional bf16 shadow for GEMM consumption.
// ---------------------------------------------------------------------------
__global__ __launch_bounds__(256)
void addln_kernel(const float* __restrict__ a, const float* __restrict__ bres,
                  const float* __restrict__ w, const float* __restrict__ bias,
                  float* __restrict__ out, __bf16* __restrict__ out16, int rows) {
  long long gid = (long long)blockIdx.x * blockDim.x + threadIdx.x;
  int lane = (int)(gid & 31);
  long long row = gid >> 5;
  if (row >= rows) return;
  int cbase = lane * 8;
  const float* pa = a + row * kC + cbase;
  float v[8];
#pragma unroll
  for (int i = 0; i < 8; ++i) v[i] = pa[i];
  if (bres) {
    const float* pb = bres + row * kC + cbase;
#pragma unroll
    for (int i = 0; i < 8; ++i) v[i] += pb[i];
  }
  float s = 0.f, s2 = 0.f;
#pragma unroll
  for (int i = 0; i < 8; ++i) { s += v[i]; s2 += v[i] * v[i]; }
#pragma unroll
  for (int off = 16; off >= 1; off >>= 1) {
    s  += __shfl_xor(s,  off, 32);
    s2 += __shfl_xor(s2, off, 32);
  }
  float mean = s * (1.0f / kC);
  float var  = s2 * (1.0f / kC) - mean * mean;
  float rs   = rsqrtf(var + 1e-5f);
  float* po = out + row * kC + cbase;
#pragma unroll
  for (int i = 0; i < 8; ++i) {
    float o = (v[i] - mean) * rs * w[cbase + i] + bias[cbase + i];
    po[i] = o;
    if (out16) out16[row * kC + cbase + i] = (__bf16)o;
  }
}

__global__ void add_kernel(const float* __restrict__ a, const float* __restrict__ b,
                           float* __restrict__ out, __bf16* __restrict__ out16,
                           long long n) {
  long long i = (long long)blockIdx.x * blockDim.x + threadIdx.x;
  if (i < n) {
    float v = a[i] + b[i];
    out[i]   = v;
    out16[i] = (__bf16)v;
  }
}

__global__ void copy_kernel(const float* __restrict__ a, float* __restrict__ out,
                            long long n) {
  long long i = (long long)blockIdx.x * blockDim.x + threadIdx.x;
  if (i < n) out[i] = a[i];
}

// ---------------------------------------------------------------------------
// Host orchestration.
// d_in order (setup_inputs insertion order, params flattened recursively in
// insertion order): 0:q0 1:kv0 2:q1 3:kv1 4:q2 5:kv2 6:q3 7:kv3 8:level_embed
// then per layer (base = 9 + 36*layer):
//   +0..7  self  {so_w, so_b, aw_w, aw_b, v_w, v_b, o_w, o_b}
//   +8..15 t1, +16..23 t2 (same order)
//   +24..31 norm0_w, norm0_b, norm1_w, norm1_b, norm2_w, norm2_b, norm5_w, norm5_b
//   +32..35 lin1_w, lin1_b, lin2_w, lin2_b
// ---------------------------------------------------------------------------
extern "C" void kernel_launch(void* const* d_in, const int* in_sizes, int n_in,
                              void* d_out, int out_size, void* d_ws, size_t ws_size,
                              hipStream_t stream) {
  (void)in_sizes; (void)n_in; (void)out_size; (void)ws_size;
  auto F = [&](int i) { return (const float*)d_in[i]; };

  const float *q0 = F(0), *q1 = F(2), *q2 = F(4), *q3 = F(6);
  const float *k0 = F(1), *k1 = F(3), *k2 = F(5), *k3 = F(7);
  const float *le = F(8);

  // fp32 region ------------------------------------------------------------
  float* f = (float*)d_ws;
  float* bufA = f;  f += kBLC;       // running query state
  float* bufB = f;  f += kBLC;       // q after norm0
  float* valB = f;  f += kBLC;       // value projection
  float* soB  = f;  f += 2 * kBLC;   // sampling offsets (B,L,512)
  float* awB  = f;  f += kBLC / 2;   // attn weights (B,L,128)
  float* msda = f;  f += kBLC;       // msda / ffn temp
  float* s2B  = f;  f += kBLC;       // src2
  float* s3B  = f;  f += kBLC;       // src3
  float* xB   = f;  f += kBLC;       // x = q + src3
  // bf16 region ------------------------------------------------------------
  __bf16* h = (__bf16*)f;
  __bf16* bufA16 = h;  h += kBLC;
  __bf16* bufB16 = h;  h += kBLC;
  __bf16* kv1p16 = h;  h += kBLC;
  __bf16* kv2p16 = h;  h += kBLC;
  __bf16* samp16 = h;  h += kBLC;
  __bf16* xB16   = h;  h += kBLC;
  __bf16* hB16   = h;  h += 4 * kBLC;   // ffn hidden (B,L,1024), bf16-only
  __bf16* wpool  = h;                   // bf16 weight pool (~4.23M elems)

  // Convert all GEMM weight matrices to bf16 once per launch.
  long long wOff = 0;
  auto cvtW = [&](int idx, long long nElems) -> const __bf16* {
    __bf16* dst = wpool + wOff;
    wOff += nElems;
    cvt_bf16_kernel<<<(int)((nElems + 255) / 256), 256, 0, stream>>>(F(idx), dst, nElems);
    return dst;
  };
  const __bf16* aw16[3][3][4];  // [layer][attn][so,aw,v,o]
  const __bf16* lw16[3][2];     // [layer][lin1,lin2]
  for (int ly = 0; ly < 3; ++ly) {
    int base = 9 + ly * 36;
    for (int at = 0; at < 3; ++at) {
      int pb = base + at * 8;
      aw16[ly][at][0] = cvtW(pb + 0, 512 * 256);   // so_w
      aw16[ly][at][1] = cvtW(pb + 2, 128 * 256);   // aw_w
      aw16[ly][at][2] = cvtW(pb + 4, 256 * 256);   // v_w
      aw16[ly][at][3] = cvtW(pb + 6, 256 * 256);   // o_w
    }
    lw16[ly][0] = cvtW(base + 32, 1024 * 256);     // lin1_w
    lw16[ly][1] = cvtW(base + 34, 256 * 1024);     // lin2_w
  }

  const long long total = kBLC;
  const int packBlocks = (int)((total + 255) / 256);

  pack_q_kernel<<<packBlocks, 256, 0, stream>>>(q0, q1, q2, q3, bufA, bufA16, total);
  pack_kv_kernel<<<packBlocks, 256, 0, stream>>>(k0, k1, k2, k3, le, kv1p16, kv2p16, total);

  const int gM = (kM + 63) / 64;
  auto gemm = [&](const __bf16* A, const __bf16* Wt, const float* bi,
                  float* outF, __bf16* outB, int N, int K, int relu) {
    gemm_bf16_wmma<<<dim3(gM, N / 64), 128, 0, stream>>>(A, Wt, bi, outF, outB,
                                                         kM, N, K, relu);
  };

  auto run_msda = [&](const __bf16* Q, const __bf16* Vsrc, int ly, int at,
                      int pb, float* outBuf) {
    gemm(Vsrc, aw16[ly][at][2], F(pb + 5), valB, nullptr, 256, 256, 0); // value
    gemm(Q,    aw16[ly][at][0], F(pb + 1), soB,  nullptr, 512, 256, 0); // offsets
    gemm(Q,    aw16[ly][at][1], F(pb + 3), awB,  nullptr, 128, 256, 0); // logits
    int groups = kM * kHeads;
    softmax16_kernel<<<(groups + 255) / 256, 256, 0, stream>>>(awB, groups);
    long long sThreads = (long long)groups * 32;
    msda_sample_kernel<<<(int)((sThreads + 255) / 256), 256, 0, stream>>>(
        valB, soB, awB, samp16);
    gemm(samp16, aw16[ly][at][3], F(pb + 7), outBuf, nullptr, 256, 256, 0); // o-proj
  };

  const int lnBlocks = (int)(((long long)kM * 32 + 255) / 256);
  auto addln = [&](const float* a, const float* bres, const float* wn,
                   const float* bn, float* out, __bf16* out16) {
    addln_kernel<<<lnBlocks, 256, 0, stream>>>(a, bres, wn, bn, out, out16, kM);
  };

  for (int layer = 0; layer < 3; ++layer) {
    int base = 9 + layer * 36;
    run_msda(bufA16, bufA16, layer, 0, base + 0, msda);              // self-attn
    addln(bufA, msda, F(base + 24), F(base + 25), bufB, bufB16);     // norm0 -> q
    run_msda(bufB16, kv1p16, layer, 1, base + 8, msda);              // cross t1
    addln(msda, nullptr, F(base + 26), F(base + 27), s2B, nullptr);  // norm1 -> src2
    run_msda(bufB16, kv2p16, layer, 2, base + 16, msda);             // cross t2
    addln(s2B, msda, F(base + 28), F(base + 29), s3B, nullptr);      // norm2 -> src3
    add_kernel<<<packBlocks, 256, 0, stream>>>(bufB, s3B, xB, xB16, total);
    gemm(xB16, lw16[layer][0], F(base + 33), nullptr, hB16, 1024, 256, 1); // lin1+relu
    gemm(hB16, lw16[layer][1], F(base + 35), msda, nullptr, 256, 1024, 0); // lin2
    addln(xB, msda, F(base + 30), F(base + 31), bufA, bufA16);       // norm5 -> q_out
  }

  float* out = (float*)d_out;
  copy_kernel<<<packBlocks, 256, 0, stream>>>(s3B, out, total);          // prev
  copy_kernel<<<packBlocks, 256, 0, stream>>>(bufA, out + total, total); // out
}
